// BaseModel_40114994544863
// MI455X (gfx1250) — matrix operand order repaired and checked by reading, not compiled, not run on hardware
//
#include <hip/hip_runtime.h>
#include <cstdint>

// ---------------------------------------------------------------------------
// Dimensions
// ---------------------------------------------------------------------------
constexpr int Bn  = 128;   // batch
constexpr int CHn = 22;    // EEG channels
constexpr int Tn  = 1000;  // time
constexpr int W0n = 300;   // spatial width
constexpr int C1n = 150;   // block channels
constexpr int CFn = 100;   // fused channels
constexpr int NCn = 4;     // classes
constexpr int TP  = 1008;  // T padded to 16  (WMMA N)
constexpr int KCH = 480;   // 450 padded to 32 (WMMA K)
constexpr int TK  = 1024;  // T padded to 32  (WMMA K for covariance)
constexpr int MP  = 112;   // 100 padded to 16 (WMMA M/N)

typedef __attribute__((ext_vector_type(16))) _Float16 v16h;
typedef __attribute__((ext_vector_type(8)))  float    v8f;

union AFrag { v16h v; _Float16 h[16]; uint32_t u[8]; };

// ---------------------------------------------------------------------------
// Kernel 0: combined spatial+pointwise weights  Wc = wp @ renorm(wt)  (150x22)
// ---------------------------------------------------------------------------
__global__ __launch_bounds__(256) void k_prep_wc(const float* __restrict__ wt,
                                                 const float* __restrict__ wp,
                                                 float* __restrict__ Wc) {
  __shared__ float sc[W0n];
  int tid = threadIdx.x;
  for (int f = tid; f < W0n; f += 256) {
    float s = 0.f;
    for (int ch = 0; ch < CHn; ++ch) { float w = wt[f * CHn + ch]; s += w * w; }
    float n = sqrtf(s);
    sc[f] = fminf(1.0f, 1.0f / fmaxf(n, 1e-12f));
  }
  __syncthreads();
  for (int idx = tid; idx < C1n * CHn; idx += 256) {
    int c = idx / CHn, ch = idx % CHn;
    float acc = 0.f;
    for (int f = 0; f < W0n; ++f) acc += wp[c * W0n + f] * wt[f * CHn + ch] * sc[f];
    Wc[idx] = acc;
  }
}

// ws (100x450) -> f16 padded to (112x480), zeros in padding
__global__ void k_prep_ws16(const float* __restrict__ ws, _Float16* __restrict__ ws16) {
  int idx = blockIdx.x * 256 + threadIdx.x;
  if (idx >= MP * KCH) return;
  int r = idx / KCH, k = idx % KCH;
  ws16[idx] = (r < CFn && k < 450) ? (_Float16)ws[r * 450 + k] : (_Float16)0.f;
}

// zero Hcat channel-padding rows 450..479 (u32-wide)
__global__ void k_zero_hcat_pad(uint32_t* __restrict__ hc) {
  int idx = blockIdx.x * 256 + threadIdx.x;
  const int perB = 30 * TP / 2;
  if (idx >= Bn * perB) return;
  int b = idx / perB, w = idx - b * perB;
  hc[(size_t)b * (KCH * TP / 2) + 450 * (TP / 2) + w] = 0u;
}

// ---------------------------------------------------------------------------
// Kernel 1: per (batch, block): h = Wc@x ; LN ; depthwise conv ; LN ; -> f16
// One wave per channel; x[b] cached in LDS; per-wave row buffers.
// ---------------------------------------------------------------------------
struct BlkP { const float* ga; const float* ba; const float* wd;
              const float* gb; const float* bb; int k; int coff; };

__global__ __launch_bounds__(256) void k_blocks(const float* __restrict__ feat,
                                                const float* __restrict__ Wc,
                                                BlkP p0, BlkP p1, BlkP p2,
                                                _Float16* __restrict__ Hcat) {
  extern __shared__ float sm[];
  float* xs  = sm;               // 22*1000
  float* r1  = xs + CHn * Tn;    // 8*1000
  float* r2  = r1 + 8 * Tn;      // 8*1000
  float* wds = r2 + 8 * Tn;      // 8*96
  int blk = blockIdx.x, b = blockIdx.y;
  BlkP p = (blk == 0) ? p0 : (blk == 1) ? p1 : p2;
  int tid = threadIdx.x, wave = tid >> 5, lane = tid & 31;

  for (int i = tid; i < CHn * Tn; i += 256) xs[i] = feat[(size_t)b * CHn * Tn + i];
  __syncthreads();

  float* rowa = r1 + wave * Tn;
  float* rowb = r2 + wave * Tn;
  float* wv   = wds + wave * 96;
  const int K = p.k, pad = (p.k - 1) >> 1;

  for (int c = wave; c < C1n; c += 8) {
    const float* w = Wc + (size_t)(blk * C1n + c) * CHn;
    float wr[CHn];
#pragma unroll
    for (int ch = 0; ch < CHn; ++ch) wr[ch] = w[ch];
    float s1 = 0.f, q1 = 0.f;
    for (int i = 0; i < 32; ++i) {
      int t = lane + 32 * i;
      if (t < Tn) {
        float a = 0.f;
#pragma unroll
        for (int ch = 0; ch < CHn; ++ch) a += wr[ch] * xs[ch * Tn + t];
        rowa[t] = a; s1 += a; q1 += a * a;
      }
    }
#pragma unroll
    for (int o = 16; o > 0; o >>= 1) { s1 += __shfl_xor(s1, o, 32); q1 += __shfl_xor(q1, o, 32); }
    float m  = s1 * (1.f / Tn);
    float inv = rsqrtf(fmaxf(q1 * (1.f / Tn) - m * m, 0.f) + 1e-5f);
    for (int i = 0; i < 32; ++i) {
      int t = lane + 32 * i;
      if (t < Tn) rowa[t] = (rowa[t] - m) * inv * p.ga[t] + p.ba[t];
    }
    for (int j = lane; j < K; j += 32) wv[j] = p.wd[(size_t)c * K + j];
    float s2 = 0.f, q2 = 0.f;
    for (int i = 0; i < 32; ++i) {
      int t = lane + 32 * i;
      if (t < Tn) {
        int j0 = pad - t;        if (j0 < 0) j0 = 0;
        int j1 = Tn + pad - t;   if (j1 > K) j1 = K;
        float a = 0.f;
        for (int j = j0; j < j1; ++j) a += wv[j] * rowa[t - pad + j];
        rowb[t] = a; s2 += a; q2 += a * a;
      }
    }
#pragma unroll
    for (int o = 16; o > 0; o >>= 1) { s2 += __shfl_xor(s2, o, 32); q2 += __shfl_xor(q2, o, 32); }
    float m2  = s2 * (1.f / Tn);
    float inv2 = rsqrtf(fmaxf(q2 * (1.f / Tn) - m2 * m2, 0.f) + 1e-5f);
    _Float16* out = Hcat + ((size_t)b * KCH + p.coff + c) * TP;
    for (int i = 0; i < 32; ++i) {
      int t = lane + 32 * i;
      if (t < Tn)      out[t] = (_Float16)((rowb[t] - m2) * inv2 * p.gb[t] + p.bb[t]);
      else if (t < TP) out[t] = (_Float16)0.f;
    }
  }
}

// ---------------------------------------------------------------------------
// Kernel 2: S[b] = ws(100x450) @ Hcat[b](450x1000)   -- WMMA f16 -> f32
// One wave per 16x16 output tile; K = 480 (15 k-steps of 32).
// ---------------------------------------------------------------------------
__global__ __launch_bounds__(256) void k_sconv(const _Float16* __restrict__ ws16,
                                               const _Float16* __restrict__ Hcat,
                                               float* __restrict__ S) {
  int wid  = blockIdx.x * 8 + (threadIdx.x >> 5);
  int lane = threadIdx.x & 31;
  int b  = wid / (7 * 63);
  int rr = wid % (7 * 63);
  int mt = rr / 63, nt = rr % 63;
  int l16 = lane & 15, half = lane >> 4;
  v8f acc = {0.f, 0.f, 0.f, 0.f, 0.f, 0.f, 0.f, 0.f};
  const _Float16* arow = ws16 + (size_t)(mt * 16 + l16) * KCH;
  const _Float16* bcol = Hcat + (size_t)b * KCH * TP + nt * 16 + l16;
  for (int kt = 0; kt < KCH / 32; ++kt) {
    AFrag a, bf;
    int kb = kt * 32;
#pragma unroll
    for (int v = 0; v < 8; ++v) {
      int ka = (v < 4) ? (2 * v + 8 * half) : (16 + 2 * (v - 4) + 8 * half);
      a.u[v] = *reinterpret_cast<const uint32_t*>(arow + kb + ka);
    }
#pragma unroll
    for (int v = 0; v < 8; ++v) {
      int kk = (v < 4) ? (2 * v + 8 * half) : (16 + 2 * (v - 4) + 8 * half);
      bf.h[2 * v]     = bcol[(size_t)(kb + kk) * TP];
      bf.h[2 * v + 1] = bcol[(size_t)(kb + kk + 1) * TP];
    }
    acc = __builtin_amdgcn_wmma_f32_16x16x32_f16(false, a.v, false, bf.v,
                                                 (short)0, acc, false, false);
  }
  int t = nt * 16 + l16;
  if (t < Tn) {
#pragma unroll
    for (int r = 0; r < 8; ++r) {
      int row = mt * 16 + r + 8 * half;
      if (row < CFn) S[((size_t)b * CFn + row) * Tn + t] = acc[r];
    }
  }
}

// ---------------------------------------------------------------------------
// Kernel 3: row means, centered f16 copy Xm (112x1024, zero-padded)
// ---------------------------------------------------------------------------
__global__ __launch_bounds__(256) void k_center(const float* __restrict__ S,
                                                _Float16* __restrict__ Xm) {
  __shared__ float red[256];
  int c = blockIdx.x, b = blockIdx.y, tid = threadIdx.x;
  _Float16* xr = Xm + ((size_t)b * MP + c) * TK;
  if (c >= CFn) {
    for (int t = tid; t < TK; t += 256) xr[t] = (_Float16)0.f;
    return;
  }
  const float* sr = S + ((size_t)b * CFn + c) * Tn;
  float s = 0.f;
  for (int t = tid; t < Tn; t += 256) s += sr[t];
  red[tid] = s; __syncthreads();
  for (int o = 128; o > 0; o >>= 1) { if (tid < o) red[tid] += red[tid + o]; __syncthreads(); }
  float mu = red[0] * (1.f / Tn);
  for (int t = tid; t < TK; t += 256)
    xr[t] = (t < Tn) ? (_Float16)(sr[t] - mu) : (_Float16)0.f;
}

// ---------------------------------------------------------------------------
// Kernel 4: covariance R = Xm @ Xm^T / (T-1)  -- WMMA, both operands from Xm
// ---------------------------------------------------------------------------
__global__ __launch_bounds__(256) void k_cov(const _Float16* __restrict__ Xm,
                                             float* __restrict__ R) {
  int wid  = blockIdx.x * 8 + (threadIdx.x >> 5);
  int lane = threadIdx.x & 31;
  int b  = wid / 49;
  int rr = wid % 49;
  int mt = rr / 7, nt = rr % 7;
  int l16 = lane & 15, half = lane >> 4;
  v8f acc = {0.f, 0.f, 0.f, 0.f, 0.f, 0.f, 0.f, 0.f};
  const _Float16* arow = Xm + ((size_t)b * MP + mt * 16 + l16) * TK;
  const _Float16* brow = Xm + ((size_t)b * MP + nt * 16 + l16) * TK;
  for (int kt = 0; kt < TK / 32; ++kt) {
    AFrag a, bf;
    int kb = kt * 32;
#pragma unroll
    for (int v = 0; v < 8; ++v) {
      int ka = (v < 4) ? (2 * v + 8 * half) : (16 + 2 * (v - 4) + 8 * half);
      a.u[v] = *reinterpret_cast<const uint32_t*>(arow + kb + ka);
    }
#pragma unroll
    for (int v = 0; v < 8; ++v)
      bf.u[v] = *reinterpret_cast<const uint32_t*>(brow + kb + ((v < 4) ? (2 * v + 8 * half) : (16 + 2 * (v - 4) + 8 * half)));
    acc = __builtin_amdgcn_wmma_f32_16x16x32_f16(false, a.v, false, bf.v,
                                                 (short)0, acc, false, false);
  }
  int col = nt * 16 + l16;
  if (col < CFn) {
#pragma unroll
    for (int r = 0; r < 8; ++r) {
      int row = mt * 16 + r + 8 * half;
      if (row < CFn) R[((size_t)b * CFn + row) * CFn + col] = acc[r] * (1.f / (Tn - 1));
    }
  }
}

// ---------------------------------------------------------------------------
// Kernel 5: covariance attention: Q=wq@R, K=wk@R, A=softmax(QK^T/10),
//           Ratt = A R A^T + eps I.  One workgroup / batch, all in LDS.
// ---------------------------------------------------------------------------
__global__ __launch_bounds__(256) void k_attn(const float* __restrict__ R,
                                              const float* __restrict__ wq,
                                              const float* __restrict__ wk,
                                              float* __restrict__ Ratt) {
  extern __shared__ float sm[];
  float* Rs = sm;
  float* Qs = Rs + 10000;
  float* Ks = Qs + 10000;
  float* As = Ks + 10000;
  int b = blockIdx.x, tid = threadIdx.x;
  for (int i = tid; i < 10000; i += 256) Rs[i] = R[(size_t)b * 10000 + i];
  __syncthreads();
  for (int idx = tid; idx < 10000; idx += 256) {
    int i = idx / 100, k = idx % 100;
    float q = 0.f, kk = 0.f;
    for (int m = 0; m < 100; ++m) {
      float r = Rs[m * 100 + k];
      q  += wq[i * 100 + m] * r;
      kk += wk[i * 100 + m] * r;
    }
    Qs[idx] = q; Ks[idx] = kk;
  }
  __syncthreads();
  for (int idx = tid; idx < 10000; idx += 256) {
    int i = idx / 100, j = idx % 100;
    float s = 0.f;
    for (int k = 0; k < 100; ++k) s += Qs[i * 100 + k] * Ks[j * 100 + k];
    As[idx] = s * 0.1f;   // / sqrt(100)
  }
  __syncthreads();
  if (tid < 100) {
    float* row = As + tid * 100;
    float mx = row[0];
    for (int j = 1; j < 100; ++j) mx = fmaxf(mx, row[j]);
    float sum = 0.f;
    for (int j = 0; j < 100; ++j) { float e = __expf(row[j] - mx); row[j] = e; sum += e; }
    float inv = 1.f / sum;
    for (int j = 0; j < 100; ++j) row[j] *= inv;
  }
  __syncthreads();
  for (int idx = tid; idx < 10000; idx += 256) {  // M1 = A @ R  (into Qs)
    int i = idx / 100, j = idx % 100;
    float s = 0.f;
    for (int k = 0; k < 100; ++k) s += As[i * 100 + k] * Rs[k * 100 + j];
    Qs[idx] = s;
  }
  __syncthreads();
  for (int idx = tid; idx < 10000; idx += 256) {  // Ratt = M1 @ A^T + eps I
    int i = idx / 100, j = idx % 100;
    float s = 0.f;
    for (int k = 0; k < 100; ++k) s += Qs[i * 100 + k] * As[j * 100 + k];
    if (i == j) s += 1e-4f;
    Ratt[(size_t)b * 10000 + idx] = s;
  }
}

// ---------------------------------------------------------------------------
// Kernel 6: batched 100x100 symmetric Jacobi eigensolver (parallel ordering),
//           L = V log(Lambda) V^T, triangular vec, FC logits.
// ---------------------------------------------------------------------------
__global__ __launch_bounds__(256) void k_logm(const float* __restrict__ Ratt,
                                              const float* __restrict__ wfc,
                                              const float* __restrict__ bfc,
                                              float* __restrict__ outLogits,
                                              float* __restrict__ outL) {
  extern __shared__ float sm[];
  const int LD = 101;
  float* A  = sm;                  // 100*101
  float* V  = A + 100 * LD;        // 100*101
  float* cs = V + 100 * LD;        // 50
  float* sn = cs + 50;             // 50
  int*   pi = (int*)(sn + 50);     // 50
  int*   pj = pi + 50;             // 50
  float* ll = (float*)(pj + 50);   // 100
  float* rb = ll + 100;            // 256*4 logit reduction buffer
  int b = blockIdx.x, tid = threadIdx.x;

  for (int idx = tid; idx < 10000; idx += 256) {
    int i = idx / 100, j = idx % 100;
    A[i * LD + j] = Ratt[(size_t)b * 10000 + idx];
    V[i * LD + j] = (i == j) ? 1.f : 0.f;
  }
  __syncthreads();

  for (int sweep = 0; sweep < 10; ++sweep) {
    for (int r = 0; r < 99; ++r) {
      if (tid < 50) {   // tournament pairing + rotation angles
        int p = tid, i, j;
        if (p == 0) { i = 99; j = r % 99; }
        else        { i = (r + p) % 99; j = (r - p + 99) % 99; }
        if (i > j) { int t = i; i = j; j = t; }
        pi[p] = i; pj[p] = j;
        float apq = A[i * LD + j];
        float c = 1.f, s = 0.f;
        if (fabsf(apq) > 1e-12f) {
          float tau = (A[j * LD + j] - A[i * LD + i]) / (2.f * apq);
          float tt  = ((tau >= 0.f) ? 1.f : -1.f) / (fabsf(tau) + sqrtf(1.f + tau * tau));
          c = rsqrtf(1.f + tt * tt);
          s = tt * c;
        }
        cs[p] = c; sn[p] = s;
      }
      __syncthreads();
      for (int idx = tid; idx < 5000; idx += 256) {  // row phase (disjoint rows)
        int p = idx / 100, k = idx % 100;
        int i = pi[p], j = pj[p];
        float c = cs[p], s = sn[p];
        float ai = A[i * LD + k], aj = A[j * LD + k];
        A[i * LD + k] = c * ai - s * aj;
        A[j * LD + k] = s * ai + c * aj;
      }
      __syncthreads();
      for (int idx = tid; idx < 5000; idx += 256) {  // col phase + eigvecs
        int p = idx / 100, k = idx % 100;
        int i = pi[p], j = pj[p];
        float c = cs[p], s = sn[p];
        float ai = A[k * LD + i], aj = A[k * LD + j];
        A[k * LD + i] = c * ai - s * aj;
        A[k * LD + j] = s * ai + c * aj;
        float vi = V[k * LD + i], vj = V[k * LD + j];
        V[k * LD + i] = c * vi - s * vj;
        V[k * LD + j] = s * vi + c * vj;
      }
      __syncthreads();
    }
  }

  if (tid < 100) ll[tid] = logf(fmaxf(A[tid * LD + tid], 1e-6f));
  __syncthreads();

  float part[4] = {0.f, 0.f, 0.f, 0.f};
  const float rt2 = 1.41421356237309515f;
  for (int idx = tid; idx < 10000; idx += 256) {
    int i = idx / 100, j = idx % 100;
    float s = 0.f;
    for (int m = 0; m < 100; ++m) s += V[i * LD + m] * ll[m] * V[j * LD + m];
    outL[(size_t)b * 10000 + idx] = s;
    if (i <= j) {
      float vecv = s * ((i == j) ? 1.f : rt2);
      int tri = i * 100 - (i * (i - 1)) / 2 + (j - i);
#pragma unroll
      for (int c = 0; c < 4; ++c) part[c] += vecv * wfc[c * 5050 + tri];
    }
  }
#pragma unroll
  for (int c = 0; c < 4; ++c) rb[tid * 4 + c] = part[c];
  __syncthreads();
  for (int o = 128; o > 0; o >>= 1) {   // deterministic tree reduction
    if (tid < o) {
#pragma unroll
      for (int c = 0; c < 4; ++c) rb[tid * 4 + c] += rb[(tid + o) * 4 + c];
    }
    __syncthreads();
  }
  if (tid < 4) outLogits[b * 4 + tid] = rb[tid] + bfc[tid];
}

// ---------------------------------------------------------------------------
// Launch
// ---------------------------------------------------------------------------
extern "C" void kernel_launch(void* const* d_in, const int* in_sizes, int n_in,
                              void* d_out, int out_size, void* d_ws, size_t ws_size,
                              hipStream_t stream) {
  (void)in_sizes; (void)n_in; (void)out_size; (void)ws_size;
  const float* feat  = (const float*)d_in[0];
  const float* wt[3] = {(const float*)d_in[1],  (const float*)d_in[9],  (const float*)d_in[17]};
  const float* wp[3] = {(const float*)d_in[2],  (const float*)d_in[10], (const float*)d_in[18]};
  // bp (3,11,19) and bs (26) provably cancel under LayerNorm / covariance centering.
  const float* ga[3] = {(const float*)d_in[4],  (const float*)d_in[12], (const float*)d_in[20]};
  const float* ba[3] = {(const float*)d_in[5],  (const float*)d_in[13], (const float*)d_in[21]};
  const float* wd[3] = {(const float*)d_in[6],  (const float*)d_in[14], (const float*)d_in[22]};
  const float* gb[3] = {(const float*)d_in[7],  (const float*)d_in[15], (const float*)d_in[23]};
  const float* bb[3] = {(const float*)d_in[8],  (const float*)d_in[16], (const float*)d_in[24]};
  const float* wsW = (const float*)d_in[25];
  const float* wq  = (const float*)d_in[27];
  const float* wk  = (const float*)d_in[28];
  const float* wfc = (const float*)d_in[29];
  const float* bfc = (const float*)d_in[30];

  // workspace layout (~215 MB)
  char* wsb = (char*)d_ws;
  size_t off = 0;
  auto take = [&](size_t bytes) -> char* {
    char* p = wsb + off;
    off = (off + bytes + 255) & ~(size_t)255;
    return p;
  };
  float*    Wc   = (float*)   take((size_t)3 * C1n * CHn * 4);
  _Float16* Hcat = (_Float16*)take((size_t)Bn * KCH * TP * 2);
  float*    S    = (float*)   take((size_t)Bn * CFn * Tn * 4);
  _Float16* Xm   = (_Float16*)take((size_t)Bn * MP * TK * 2);
  float*    R    = (float*)   take((size_t)Bn * CFn * CFn * 4);
  float*    Ra   = (float*)   take((size_t)Bn * CFn * CFn * 4);
  _Float16* ws16 = (_Float16*)take((size_t)MP * KCH * 2);

  for (int blk = 0; blk < 3; ++blk)
    k_prep_wc<<<1, 256, 0, stream>>>(wt[blk], wp[blk], Wc + blk * C1n * CHn);
  k_prep_ws16<<<(MP * KCH + 255) / 256, 256, 0, stream>>>(wsW, ws16);
  k_zero_hcat_pad<<<(Bn * 30 * TP / 2 + 255) / 256, 256, 0, stream>>>((uint32_t*)Hcat);

  BlkP p0{ga[0], ba[0], wd[0], gb[0], bb[0], 15, 0};
  BlkP p1{ga[1], ba[1], wd[1], gb[1], bb[1], 95, 150};
  BlkP p2{ga[2], ba[2], wd[2], gb[2], bb[2], 55, 300};
  size_t sm1 = (size_t)(CHn * Tn + 16 * Tn + 8 * 96) * 4;   // 155072 B
  k_blocks<<<dim3(3, Bn), 256, sm1, stream>>>(feat, Wc, p0, p1, p2, Hcat);

  k_sconv<<<(Bn * 7 * 63) / 8, 256, 0, stream>>>(ws16, Hcat, S);
  k_center<<<dim3(MP, Bn), 256, 0, stream>>>(S, Xm);
  k_cov<<<(Bn * 49) / 8, 256, 0, stream>>>(Xm, R);
  k_attn<<<Bn, 256, (size_t)40000 * 4, stream>>>(R, wq, wk, Ra);

  size_t sm6 = (size_t)(100 * 101 * 2 + 50 * 4 + 100 + 256 * 4) * 4;  // ~86 KB
  k_logm<<<Bn, 256, sm6, stream>>>(Ra, wfc, bfc, (float*)d_out,
                                   (float*)d_out + (size_t)Bn * NCn);
}
